// Encoding_15290083573959
// MI455X (gfx1250) — matrix operand order repaired and checked by reading, not compiled, not run on hardware
//
#include <hip/hip_runtime.h>

// Problem constants (from the reference)
#define D_DIM   128      // feature dim
#define K_DIM   32       // codewords
#define B_DIM   32       // batch
#define N_TOT   9216     // H*W = 96*96
#define TILE_N  128      // spatial positions per block
#define DPAD    132      // padded bf16 row stride for X tile in LDS (even, 264B rows)
#define THREADS 256      // 8 waves of 32 (wave32)

typedef __attribute__((ext_vector_type(16))) __bf16 v16bf;
typedef __attribute__((ext_vector_type(8)))  float  v8f;

union FragBF { v16bf v; unsigned int u[8]; };

__device__ __forceinline__ unsigned short f2bf(float f) {
    // round-to-nearest-even fp32 -> bf16
    unsigned int b = __float_as_uint(f);
    unsigned int r = b + 0x7FFFu + ((b >> 16) & 1u);
    return (unsigned short)(r >> 16);
}

__device__ __forceinline__ unsigned int lds_addr(const void* p) {
    // Generic shared-space pointer: addr[31:0] is the LDS offset (ISA 10.2).
    return (unsigned int)(unsigned long long)p;
}

__global__ void __launch_bounds__(THREADS)
enc_zero_kernel(float* __restrict__ p, int n) {
    int i = blockIdx.x * blockDim.x + threadIdx.x;
    if (i < n) p[i] = 0.0f;
}

__global__ void __launch_bounds__(THREADS)
enc_fused_kernel(const float* __restrict__ Xg,
                 const float* __restrict__ Cw,
                 const float* __restrict__ scale,
                 float* __restrict__ out) {
    __shared__ __align__(16) unsigned short XbfS[TILE_N * DPAD]; // X tile, bf16, [n][d]
    __shared__ __align__(16) unsigned short CbS[K_DIM * D_DIM];  // codewords, bf16, [k][d]
    __shared__ __align__(16) float xcS[TILE_N * 33];             // staging / xc / A (bf16)
    __shared__ float x2S[TILE_N];
    __shared__ float colsumS[K_DIM];
    __shared__ float scale2S[K_DIM];
    __shared__ float c2S[K_DIM];

    const int tid  = threadIdx.x;
    const int lane = tid & 31;
    const int wave = tid >> 5;     // 0..7
    const int h    = lane >> 4;    // half-wave select (CDNA5 wave32 WMMA striping)
    const int l16  = lane & 15;

    const int tilesPerB = N_TOT / TILE_N;            // 72
    const int b  = blockIdx.x / tilesPerB;
    const int n0 = (blockIdx.x % tilesPerB) * TILE_N;

    // ---------- Phase 0a: async-copy codewords (fp32, 16KB) into LDS staging ----------
    // Staging buffer = xcS (idle until GEMM1). One b128 async copy per 16B chunk.
    {
        const float*  gsrc  = Cw + tid * 4;                       // 16B per copy
        unsigned int  lbase = lds_addr(&xcS[tid * 4]);
        #pragma unroll
        for (int i = 0; i < (K_DIM * D_DIM) / (THREADS * 4); ++i) {   // 4 copies/thread
            const float* g = gsrc + i * (THREADS * 4);
            unsigned int l = lbase + i * (THREADS * 16);
            asm volatile("global_load_async_to_lds_b128 %0, %1, off"
                         :: "v"(l), "v"(g) : "memory");
        }
    }
    if (tid < K_DIM) {
        float s = scale[tid];
        scale2S[tid] = s * s;
        colsumS[tid] = 0.0f;
    }
    if (tid < TILE_N) x2S[tid] = 0.0f;
    asm volatile("s_wait_asynccnt 0x0" ::: "memory");   // my wave's copies landed
    __syncthreads();                                    // everyone's copies visible

    // ---------- Phase 0b: convert codewords to bf16, c2[k] from LDS ----------
    #pragma unroll
    for (int i = 0; i < (K_DIM * D_DIM) / (THREADS * 2); ++i) {   // 8 pairs/thread
        int p = tid + i * THREADS;                                // pair index
        float f0 = xcS[p * 2], f1 = xcS[p * 2 + 1];
        *(unsigned int*)&CbS[p * 2] =
            (unsigned int)f2bf(f0) | ((unsigned int)f2bf(f1) << 16);
    }
    if (tid < K_DIM) {
        float acc = 0.0f;
        #pragma unroll 8
        for (int d = 0; d < D_DIM; ++d) { float c = xcS[tid * D_DIM + d]; acc += c * c; }
        c2S[tid] = acc;
    }
    __syncthreads();   // CbS/c2 ready; staging may now be overwritten

    // ---------- Phase 1: load X tile (transpose [d][n] -> [n][d]), x2 on the fly ----------
    // Two d-values packed per ds_store_b32; global side stays fully coalesced over n.
    {
        const int nIdx  = tid & (TILE_N - 1);
        const int dBase = (tid >> 7) * 2;            // 0 or 2
        const float* src = Xg + (size_t)b * D_DIM * N_TOT + n0 + nIdx;
        float sq = 0.0f;
        #pragma unroll 4
        for (int d0 = dBase; d0 < D_DIM; d0 += 4) {
            float v0 = src[(size_t)d0       * N_TOT];
            float v1 = src[(size_t)(d0 + 1) * N_TOT];
            if (d0 + 4 < D_DIM)
                __builtin_prefetch(&src[(size_t)(d0 + 4) * N_TOT], 0, 0);
            sq += v0 * v0 + v1 * v1;
            *(unsigned int*)&XbfS[nIdx * DPAD + d0] =
                (unsigned int)f2bf(v0) | ((unsigned int)f2bf(v1) << 16);
        }
        atomicAdd(&x2S[nIdx], sq);                   // ds_add_f32
    }
    __syncthreads();

    // ---------- Phase 2: GEMM1  xc[n,k] = sum_d X[n,d] * C[k,d]  (bf16 WMMA) ----------
    {
        const int mrow = wave * 16 + l16;            // local n row of this lane's A stripe
        v8f acc0 = {0.f,0.f,0.f,0.f,0.f,0.f,0.f,0.f};
        v8f acc1 = {0.f,0.f,0.f,0.f,0.f,0.f,0.f,0.f};
        #pragma unroll
        for (int dc = 0; dc < D_DIM; dc += 32) {
            FragBF a, b0, b1;
            #pragma unroll
            for (int v = 0; v < 8; ++v) {
                // 16-bit A 16x32 layout: K = (v<4?0:16) + (v&3)*2 + h*8
                int d0 = dc + ((v & 4) ? 16 : 0) + (v & 3) * 2 + h * 8;
                a.u[v] = *(const unsigned int*)&XbfS[mrow * DPAD + d0];
            }
            #pragma unroll
            for (int v = 0; v < 8; ++v) {
                // 16-bit B 32x16 layout: K = (v<4?0:16) + (v&3)*2 + h*8
                int d0 = dc + ((v & 4) ? 16 : 0) + (v & 3) * 2 + h * 8;
                b0.u[v] = *(const unsigned int*)&CbS[(l16)      * D_DIM + d0];
                b1.u[v] = *(const unsigned int*)&CbS[(16 + l16) * D_DIM + d0];
            }
            acc0 = __builtin_amdgcn_wmma_f32_16x16x32_bf16(false, a.v, false, b0.v,
                                                           (short)0, acc0, false, false);
            acc1 = __builtin_amdgcn_wmma_f32_16x16x32_bf16(false, a.v, false, b1.v,
                                                           (short)0, acc1, false, false);
        }
        // f32 C/D layout: VGPR r holds row M = r + 8*h, column N = l16
        #pragma unroll
        for (int r = 0; r < 8; ++r) {
            int row = wave * 16 + r + 8 * h;
            xcS[row * 33 + l16]      = acc0[r];
            xcS[row * 33 + 16 + l16] = acc1[r];
        }
    }
    __syncthreads();

    // ---------- Phase 3: fp32 softmax over k; write A (bf16) aliased onto xcS ----------
    unsigned short* Ab = (unsigned short*)xcS;       // row stride 66 ushorts (= 33 floats)
    if (tid < TILE_N) {
        const float x2 = x2S[tid];
        float sl[K_DIM];
        float mx = -3.4e38f;
        #pragma unroll
        for (int k = 0; k < K_DIM; ++k) {
            float xc = xcS[tid * 33 + k];
            float v  = scale2S[k] * (x2 - 2.0f * xc + c2S[k]);
            sl[k] = v;
            mx = fmaxf(mx, v);
        }
        float sum = 0.0f;
        #pragma unroll
        for (int k = 0; k < K_DIM; ++k) { float e = __expf(sl[k] - mx); sl[k] = e; sum += e; }
        float inv = 1.0f / sum;
        #pragma unroll
        for (int k = 0; k < K_DIM; ++k) {
            float a = sl[k] * inv;
            Ab[tid * 66 + k] = f2bf(a);              // overwrite this row only (read first)
            atomicAdd(&colsumS[k], a);
        }
    }
    __syncthreads();

    // ---------- Phase 4: GEMM2  E[k,d] += sum_n A[n,k]*X[n,d]  - colsum[k]*C[k,d] ----------
    {
        const int dt   = wave;                       // each wave owns one 16-wide d tile
        const int dcol = dt * 16 + l16;
        #pragma unroll
        for (int kt = 0; kt < 2; ++kt) {
            const int kRow = kt * 16 + l16;
            v8f acc = {0.f,0.f,0.f,0.f,0.f,0.f,0.f,0.f};
            #pragma unroll
            for (int nc = 0; nc < TILE_N; nc += 32) {
                FragBF a, bb;
                #pragma unroll
                for (int v = 0; v < 8; ++v) {        // A^T[k,n] fragment (transposed LDS read)
                    int nb = nc + ((v & 4) ? 16 : 0) + (v & 3) * 2 + h * 8;
                    a.u[v] = (unsigned int)Ab[nb * 66 + kRow]
                           | ((unsigned int)Ab[(nb + 1) * 66 + kRow] << 16);
                }
                #pragma unroll
                for (int v = 0; v < 8; ++v) {        // B = X[n,d] fragment
                    int nb = nc + ((v & 4) ? 16 : 0) + (v & 3) * 2 + h * 8;
                    bb.u[v] = (unsigned int)XbfS[nb * DPAD + dcol]
                            | ((unsigned int)XbfS[(nb + 1) * DPAD + dcol] << 16);
                }
                acc = __builtin_amdgcn_wmma_f32_16x16x32_bf16(false, a.v, false, bb.v,
                                                              (short)0, acc, false, false);
            }
            // Fold -colsum[k]*C[k,d] into the per-block partial and reduce to global.
            #pragma unroll
            for (int r = 0; r < 8; ++r) {
                int k = kt * 16 + r + 8 * h;
                float val = acc[r] - colsumS[k] * Cw[k * D_DIM + dcol];
                atomicAdd(&out[((size_t)b * K_DIM + k) * D_DIM + dcol], val);
            }
        }
    }
}

extern "C" void kernel_launch(void* const* d_in, const int* in_sizes, int n_in,
                              void* d_out, int out_size, void* d_ws, size_t ws_size,
                              hipStream_t stream) {
    const float* X     = (const float*)d_in[0];   // [B, D, H, W] fp32
    const float* Cw    = (const float*)d_in[1];   // [K, D] fp32
    const float* scale = (const float*)d_in[2];   // [K] fp32
    float* out = (float*)d_out;                   // [B, K, D] fp32

    // Output is accumulated with atomics across 2304 partial blocks: zero it first.
    enc_zero_kernel<<<(out_size + THREADS - 1) / THREADS, THREADS, 0, stream>>>(out, out_size);

    const int grid = B_DIM * (N_TOT / TILE_N);    // 32 * 72 = 2304 blocks
    enc_fused_kernel<<<grid, THREADS, 0, stream>>>(X, Cw, scale, out);
}
